// LocationAwareAttention_87832081203829
// MI455X (gfx1250) — matrix operand order repaired and checked by reading, not compiled, not run on hardware
//
#include <hip/hip_runtime.h>
#include <hip/hip_bf16.h>

#define B_   32
#define T_   1500
#define D_   512
#define A_   512
#define C_   32
#define FS_  15
#define KW_  31

typedef __attribute__((ext_vector_type(16))) __bf16 bf16x16;
typedef __attribute__((ext_vector_type(8)))  __bf16 bf16x8;
typedef __attribute__((ext_vector_type(8)))  float  f32x8;

// ---------------------------------------------------------------------------
// Prep: W_enc (D,A) -> bf16 Wt (A,D);  W_att (C,A) -> bf16 Wat (A,C)
// ---------------------------------------------------------------------------
__global__ __launch_bounds__(256) void prep_w_kernel(
    const float* __restrict__ W_enc, const float* __restrict__ W_att,
    __bf16* __restrict__ Wt, __bf16* __restrict__ Wat) {
  int i = blockIdx.x * 256 + threadIdx.x;
  if (i < D_ * A_) {
    int d = i / A_, a = i % A_;
    Wt[(size_t)a * D_ + d] = (__bf16)W_enc[i];
  }
  if (i < C_ * A_) {
    int c = i / A_, a = i % A_;
    Wat[a * C_ + c] = (__bf16)W_att[i];
  }
}

// ---------------------------------------------------------------------------
// Prep: addv[b][a] = dec_z[b]·W_dec[:,a] + b_enc[a]
// ---------------------------------------------------------------------------
__global__ __launch_bounds__(512) void prep_dec_kernel(
    const float* __restrict__ dec_z, const float* __restrict__ W_dec,
    const float* __restrict__ b_enc, float* __restrict__ addv) {
  int b = blockIdx.x;
  int a = threadIdx.x;
  float s = b_enc[a];
  for (int d = 0; d < D_; ++d)
    s += dec_z[b * D_ + d] * W_dec[d * A_ + a];
  addv[b * A_ + a] = s;
}

// ---------------------------------------------------------------------------
// Fused: E[b][t] = w_g · tanh( enc@W_enc + conv(att_prev)@W_att + addv ) + b_g
// One block = (b, 16-row t-tile). 8 waves, each owns 4 N-tiles of 16 cols.
// ---------------------------------------------------------------------------
__global__ __launch_bounds__(256) void fused_e_kernel(
    const float* __restrict__ enc,        // (B,T,D)
    const float* __restrict__ att_prev,   // (B,T)
    const float* __restrict__ conv_w,     // (C,1,KW)
    const __bf16* __restrict__ Wt,        // (A,D) bf16
    const __bf16* __restrict__ Wat,       // (A,C) bf16
    const float* __restrict__ addv,       // (B,A)
    const float* __restrict__ w_g,        // (A)
    const float* __restrict__ b_g,        // scalar
    float* __restrict__ E)                // (B,T)
{
  __shared__ __bf16 encs[16][D_];   // 16 KB
  __shared__ float  convs[16][C_];  // 2 KB
  __shared__ float  ered[8][16];

  const int b   = blockIdx.y;
  const int t0  = blockIdx.x * 16;
  const int tid = threadIdx.x;
  const int wave = tid >> 5;
  const int lane = tid & 31;

  // Stage encoder tile as bf16 (rows past T zero-filled).
  for (int i = tid; i < 16 * D_; i += 256) {
    int r = i >> 9, col = i & (D_ - 1);
    int t = t0 + r;
    float v = (t < T_) ? enc[((size_t)b * T_ + t) * D_ + col] : 0.f;
    encs[r][col] = (__bf16)v;
  }
  // Location conv tile: convs[r][c] = sum_k conv_w[c][k] * att_prev[b][t+k-FS]
  for (int i = tid; i < 16 * C_; i += 256) {
    int r = i >> 5, c = i & 31;
    int t = t0 + r;
    float s = 0.f;
    if (t < T_) {
      #pragma unroll
      for (int k = 0; k < KW_; ++k) {
        int p = t + k - FS_;
        float ap = (p >= 0 && p < T_) ? att_prev[b * T_ + p] : 0.f;
        s += conv_w[c * KW_ + k] * ap;
      }
    }
    convs[r][c] = s;
  }
  __syncthreads();

  // WMMA k order chosen by this code, shared by A and B (wave32):
  //  A (16xK, 16-bit): lane m=lane&15; lanes 0-15 hold K 0..15,
  //                    lanes 16-31 hold K 16..31 (contiguous per lane).
  //  B (Kx16, 16-bit): lane n=lane&15; lanes 0-15 hold K 0..15,
  //                    lanes 16-31 hold K 16..31 (contiguous per lane).
  const int m     = lane & 15;
  const int hi    = lane >> 4;
  const int aoff  = hi * 16;
  const int boff  = hi * 16;

  f32x8 acc[4];
  #pragma unroll
  for (int ct = 0; ct < 4; ++ct)
    #pragma unroll
    for (int j = 0; j < 8; ++j) acc[ct][j] = 0.f;

  // Main GEMM over D in K=32 steps.
  for (int k0 = 0; k0 < D_; k0 += 32) {
    bf16x8 a0 = *(const bf16x8*)&encs[m][k0 + aoff];
    bf16x8 a1 = *(const bf16x8*)&encs[m][k0 + aoff + 8];
    bf16x16 afrag = __builtin_shufflevector(a0, a1,
        0, 1, 2, 3, 4, 5, 6, 7, 8, 9, 10, 11, 12, 13, 14, 15);
    #pragma unroll
    for (int ct = 0; ct < 4; ++ct) {
      int n = (wave * 4 + ct) * 16 + m;
      bf16x16 bfrag = *(const bf16x16*)&Wt[(size_t)n * D_ + k0 + boff];
      acc[ct] = __builtin_amdgcn_wmma_f32_16x16x32_bf16(
          false, afrag, false, bfrag, (short)0, acc[ct], false, false);
    }
  }

  // Fold conv-projection: one extra K=32 step (C==32).
  {
    bf16x16 ca;
    #pragma unroll
    for (int j = 0; j < 8; ++j) {
      ca[j]     = (__bf16)convs[m][aoff + j];
      ca[j + 8] = (__bf16)convs[m][aoff + 8 + j];
    }
    #pragma unroll
    for (int ct = 0; ct < 4; ++ct) {
      int n = (wave * 4 + ct) * 16 + m;
      bf16x16 bfrag = *(const bf16x16*)&Wat[n * C_ + boff];
      acc[ct] = __builtin_amdgcn_wmma_f32_16x16x32_bf16(
          false, ca, false, bfrag, (short)0, acc[ct], false, false);
    }
  }

  // Epilogue: e_partial[row] = sum_n tanh(acc + addv[n]) * w_g[n]
  // C/D layout: element (row = r + 8*hi, col n = tile*16 + m) in acc[.][r].
  float s[8];
  #pragma unroll
  for (int r = 0; r < 8; ++r) s[r] = 0.f;
  #pragma unroll
  for (int ct = 0; ct < 4; ++ct) {
    int n = (wave * 4 + ct) * 16 + m;
    float av = addv[b * A_ + n];
    float g  = w_g[n];
    #pragma unroll
    for (int r = 0; r < 8; ++r)
      s[r] += tanhf(acc[ct][r] + av) * g;
  }
  // Reduce over the 16 columns (lanes within each half) — deterministic.
  #pragma unroll
  for (int msk = 1; msk < 16; msk <<= 1)
    #pragma unroll
    for (int r = 0; r < 8; ++r)
      s[r] += __shfl_xor(s[r], msk, 32);
  if (m == 0) {
    #pragma unroll
    for (int r = 0; r < 8; ++r) ered[wave][hi * 8 + r] = s[r];
  }
  __syncthreads();

  if (tid < 16) {
    int t = t0 + tid;
    if (t < T_) {
      float e = b_g[0];
      #pragma unroll
      for (int w = 0; w < 8; ++w) e += ered[w][tid];
      E[b * T_ + t] = e;
    }
  }
}

// ---------------------------------------------------------------------------
// Masked softmax over T with scaling 2.0: w = softmax(2*e) (invalid t -> 0)
// ---------------------------------------------------------------------------
__global__ __launch_bounds__(256) void softmax_kernel(
    const float* __restrict__ E, const int* __restrict__ lens,
    float* __restrict__ wout) {
  __shared__ float red[256];
  int b = blockIdx.x, tid = threadIdx.x;
  int len = lens[b];
  if (len > T_) len = T_;

  float mx = -1e30f;
  for (int t = tid; t < len; t += 256) mx = fmaxf(mx, E[b * T_ + t]);
  red[tid] = mx;
  __syncthreads();
  for (int s2 = 128; s2 > 0; s2 >>= 1) {
    if (tid < s2) red[tid] = fmaxf(red[tid], red[tid + s2]);
    __syncthreads();
  }
  mx = red[0];
  __syncthreads();

  float sum = 0.f;
  for (int t = tid; t < len; t += 256)
    sum += __expf(2.f * (E[b * T_ + t] - mx));
  red[tid] = sum;
  __syncthreads();
  for (int s2 = 128; s2 > 0; s2 >>= 1) {
    if (tid < s2) red[tid] += red[tid + s2];
    __syncthreads();
  }
  float inv = 1.f / red[0];

  for (int t = tid; t < T_; t += 256) {
    float wv = (t < len) ? __expf(2.f * (E[b * T_ + t] - mx)) * inv : 0.f;
    wout[b * T_ + t] = wv;
  }
}

// ---------------------------------------------------------------------------
// Context: c[b][d] = sum_t enc[b][t][d] * w[b][t]
// ---------------------------------------------------------------------------
__global__ __launch_bounds__(128) void context_kernel(
    const float* __restrict__ enc, const float* __restrict__ wv,
    float* __restrict__ cout) {
  int b = blockIdx.y;
  int d = blockIdx.x * 128 + threadIdx.x;
  const float* er = enc + (size_t)b * T_ * D_ + d;
  const float* wr = wv + b * T_;
  float s = 0.f;
  for (int t = 0; t < T_; ++t)
    s += er[(size_t)t * D_] * wr[t];
  cout[b * D_ + d] = s;
}

// ---------------------------------------------------------------------------
extern "C" void kernel_launch(void* const* d_in, const int* in_sizes, int n_in,
                              void* d_out, int out_size, void* d_ws, size_t ws_size,
                              hipStream_t stream) {
  const float* enc      = (const float*)d_in[0];
  const float* dec_z    = (const float*)d_in[1];
  const float* att_prev = (const float*)d_in[2];
  const float* W_enc    = (const float*)d_in[3];
  const float* b_enc    = (const float*)d_in[4];
  const float* W_dec    = (const float*)d_in[5];
  const float* W_att    = (const float*)d_in[6];
  const float* conv_w   = (const float*)d_in[7];
  const float* w_g      = (const float*)d_in[8];
  const float* b_g      = (const float*)d_in[9];
  const int*   lens     = (const int*)d_in[10];

  // Workspace layout (bytes):
  //   [0)        Wt  bf16 (A,D)   = 524288
  //   [524288)   Wat bf16 (A,C)   =  32768
  //   [557056)   addv f32 (B,A)   =  65536
  //   [622592)   E    f32 (B,T)   = 192000
  char* ws = (char*)d_ws;
  __bf16* Wt   = (__bf16*)(ws);
  __bf16* Wat  = (__bf16*)(ws + 524288);
  float*  addv = (float*) (ws + 557056);
  float*  E    = (float*) (ws + 622592);

  float* c_out = (float*)d_out;            // (B, D)
  float* w_out = c_out + B_ * D_;          // (B, T)

  prep_w_kernel<<<(D_ * A_) / 256, 256, 0, stream>>>(W_enc, W_att, Wt, Wat);
  prep_dec_kernel<<<B_, 512, 0, stream>>>(dec_z, W_dec, b_enc, addv);

  dim3 ge((T_ + 15) / 16, B_);
  fused_e_kernel<<<ge, 256, 0, stream>>>(enc, att_prev, conv_w, Wt, Wat,
                                         addv, w_g, b_g, E);

  softmax_kernel<<<B_, 256, 0, stream>>>(E, lens, w_out);

  dim3 gc(D_ / 128, B_);
  context_kernel<<<gc, 128, 0, stream>>>(enc, w_out, c_out);
}
